// Modelmini_59090160058437
// MI455X (gfx1250) — matrix operand order repaired and checked by reading, not compiled, not run on hardware
//
#include <hip/hip_runtime.h>
#include <hip/hip_bf16.h>

// ---------------------------------------------------------------------------
// Problem constants (match reference)
// ---------------------------------------------------------------------------
#define NNODES 50000
#define NEDGES 800000
#define DDIM   256
#define SSAMP  5
#define NS_    (NNODES * SSAMP)      // 250000
#define TEMP_INV 2.0f                // 1/0.5
#define LAM_   0.001f
#define BN_EPS_ 1e-5f

typedef __attribute__((ext_vector_type(16))) __bf16 v16bf;
typedef __attribute__((ext_vector_type(8)))  __bf16 v8bf;
typedef __attribute__((ext_vector_type(8)))  float  v8f;

// ---------------------------------------------------------------------------
// Weight conversion with transpose: dst[n*256 + k] = (bf16) src[k*256 + n]
// One-time, 256 KB -> 128 KB per matrix; stays L2-hot for all GEMM blocks.
// ---------------------------------------------------------------------------
__global__ __launch_bounds__(256)
void f32_to_bf16_T_kernel(const float* __restrict__ src, __bf16* __restrict__ dst) {
    int k = blockIdx.x;      // 0..255
    int n = threadIdx.x;     // 0..255  (coalesced read along n)
    dst[(size_t)n * DDIM + k] = (__bf16)src[(size_t)k * DDIM + n];
}

// ---------------------------------------------------------------------------
// Degree counting + rsqrt normalizers
// ---------------------------------------------------------------------------
__global__ __launch_bounds__(256)
void deg_count_kernel(const int* __restrict__ src, const int* __restrict__ dst,
                      float* __restrict__ deg_out, float* __restrict__ deg_in, int E) {
    int e = blockIdx.x * 256 + threadIdx.x;
    if (e < E) {
        atomicAdd(&deg_out[src[e]], 1.0f);
        atomicAdd(&deg_in[dst[e]], 1.0f);
    }
}

__global__ __launch_bounds__(256)
void deg_inv_kernel(float* __restrict__ d, int n) {
    int i = blockIdx.x * 256 + threadIdx.x;
    if (i < n) {
        float v = d[i];
        d[i] = (v > 0.0f) ? rsqrtf(v) : 0.0f;   // deg>=1 when >0, max(deg,1)==deg
    }
}

// ---------------------------------------------------------------------------
// WMMA GEMM:  C[M x 256] = (rowscale? rs[m]*A[m,:] : A[m,:]) @ W[256x256] (+bias)
// W is passed TRANSPOSED (WbT[n][k], bf16). Block tile 64x256, K-step 32
// (v_wmma_f32_16x16x32_bf16), 8 waves (2x4), each wave a 32x64 sub-tile.
// LDS tiles padded to 40 elems/row (80B = 20 dwords): bank-conflict-free for
// 16-lane fragment reads AND 16B-aligned so everything moves as b128.
// ---------------------------------------------------------------------------
template<bool A_BF16, bool OUT_BF16, bool BIAS, bool RSCALE>
__global__ __launch_bounds__(256)
void gemm256_kernel(const void* __restrict__ Av, const float* __restrict__ rs,
                    const __bf16* __restrict__ WbT, const float* __restrict__ bias,
                    void* __restrict__ Cv, int M)
{
    constexpr int BM = 64, BN = 256, BK = 32, KDIM = 256, LDT = BK + 8; // 40
    __shared__ alignas(16) __bf16 As[BM][LDT];        // [m][k]
    __shared__ alignas(16) __bf16 Bs[BN][LDT];        // [n][k]  (from W^T)

    const int tid  = threadIdx.x;
    const int wave = tid >> 5;
    const int lane = tid & 31;
    const int m0   = blockIdx.x * BM;
    const int wm   = (wave & 1) * 32;     // wave's M offset: 0 / 32
    const int wn   = (wave >> 1) * 64;    // wave's N offset: 0 / 64 / 128 / 192

    // staging coordinates (uniform per thread across K iterations)
    const int ar  = tid >> 2;             // A row 0..63
    const int ac  = (tid & 3) * 8;        // A col 0,8,16,24
    const int arow_g = m0 + ar;
    const bool aok   = (arow_g < M);
    const float ascale = RSCALE ? (aok ? rs[arow_g] : 0.0f) : 1.0f;

    v8f acc[2][4] = {};

    for (int k0 = 0; k0 < KDIM; k0 += BK) {
        // ---- stage A tile (64x32): b128 loads -> one ds_store_b128 ----
        {
            v8bf av = {};
            if (A_BF16) {
                const __bf16* p = (const __bf16*)Av + (size_t)arow_g * KDIM + k0 + ac;
                __builtin_prefetch(p + BK, 0, 3);
                if (aok) av = *(const v8bf*)p;
            } else {
                const float* p = (const float*)Av + (size_t)arow_g * KDIM + k0 + ac;
                __builtin_prefetch(p + BK, 0, 3);
                float4 f0 = {}, f1 = {};
                if (aok) {
                    f0 = *(const float4*)p;
                    f1 = *(const float4*)(p + 4);
                }
                if (RSCALE) {
                    f0.x *= ascale; f0.y *= ascale; f0.z *= ascale; f0.w *= ascale;
                    f1.x *= ascale; f1.y *= ascale; f1.z *= ascale; f1.w *= ascale;
                }
                av[0] = (__bf16)f0.x; av[1] = (__bf16)f0.y;
                av[2] = (__bf16)f0.z; av[3] = (__bf16)f0.w;
                av[4] = (__bf16)f1.x; av[5] = (__bf16)f1.y;
                av[6] = (__bf16)f1.z; av[7] = (__bf16)f1.w;
            }
            *(v8bf*)&As[ar][ac] = av;
        }
        // ---- stage B tile (256 cols x 32 k) from W^T: 4x b128 both ways ----
        {
            const v8bf* p = (const v8bf*)(WbT + (size_t)tid * KDIM + k0);
            v8bf* q = (v8bf*)&Bs[tid][0];
            q[0] = p[0]; q[1] = p[1]; q[2] = p[2]; q[3] = p[3];
        }
        __syncthreads();

        // ---- build fragments per ISA 7.12.2 layouts (2x ds_load_b128 each) ----
        v16bf afrag[2], bfrag[4];
        {
            const int arow = lane & 15;
            const int kb   = (lane & 16) ? 8 : 0;  // lanes<16: K 0..7/16..23 ; >=16: 8..15/24..31
            #pragma unroll
            for (int mt = 0; mt < 2; ++mt) {
                const __bf16* p = &As[wm + mt * 16 + arow][0];
                v8bf lo = *(const v8bf*)(p + kb);
                v8bf hi = *(const v8bf*)(p + kb + 16);
                afrag[mt] = __builtin_shufflevector(lo, hi,
                    0, 1, 2, 3, 4, 5, 6, 7, 8, 9, 10, 11, 12, 13, 14, 15);
            }
            const int bk = (lane & 16) ? 8 : 0;    // lanes<16: K 0..7/16..23 ; >=16: 8..15/24..31
            #pragma unroll
            for (int nt = 0; nt < 4; ++nt) {
                const __bf16* p = &Bs[wn + nt * 16 + (lane & 15)][bk];
                v8bf lo = *(const v8bf*)p;
                v8bf hi = *(const v8bf*)(p + 16);
                bfrag[nt] = __builtin_shufflevector(lo, hi,
                    0, 1, 2, 3, 4, 5, 6, 7, 8, 9, 10, 11, 12, 13, 14, 15);
            }
        }
        #pragma unroll
        for (int mt = 0; mt < 2; ++mt)
            #pragma unroll
            for (int nt = 0; nt < 4; ++nt)
                acc[mt][nt] = __builtin_amdgcn_wmma_f32_16x16x32_bf16(
                    false, afrag[mt], false, bfrag[nt],
                    (short)0, acc[mt][nt], false, false);
        __syncthreads();
    }

    // ---- epilogue: C layout VGPR r -> M=r (lanes<16) / r+8 (lanes>=16) ----
    const int rbase = m0 + wm + ((lane & 16) ? 8 : 0);
    const int cbase = wn + (lane & 15);
    if (m0 + BM <= M) {
        // full tile: no per-element bounds checks
        #pragma unroll
        for (int mt = 0; mt < 2; ++mt) {
            #pragma unroll
            for (int nt = 0; nt < 4; ++nt) {
                #pragma unroll
                for (int r = 0; r < 8; ++r) {
                    int row = rbase + mt * 16 + r;
                    int col = cbase + nt * 16;
                    float v = acc[mt][nt][r];
                    if (BIAS) v += bias[col];
                    if (OUT_BF16) ((__bf16*)Cv)[(size_t)row * BN + col] = (__bf16)v;
                    else          ((float*)Cv)[(size_t)row * BN + col]  = v;
                }
            }
        }
    } else {
        #pragma unroll
        for (int mt = 0; mt < 2; ++mt) {
            #pragma unroll
            for (int nt = 0; nt < 4; ++nt) {
                #pragma unroll
                for (int r = 0; r < 8; ++r) {
                    int row = rbase + mt * 16 + r;
                    int col = cbase + nt * 16;
                    if (row < M) {
                        float v = acc[mt][nt][r];
                        if (BIAS) v += bias[col];
                        if (OUT_BF16) ((__bf16*)Cv)[(size_t)row * BN + col] = (__bf16)v;
                        else          ((float*)Cv)[(size_t)row * BN + col]  = v;
                    }
                }
            }
        }
    }
}

// ---------------------------------------------------------------------------
// Edge aggregation: agg[dst[e], :] += h[src[e], :]  (64 lanes/edge, float4)
// agg (51 MB) is L2-resident on the 192 MB L2 -> L2 atomics.
// ---------------------------------------------------------------------------
__global__ __launch_bounds__(256)
void edge_agg_kernel(const int* __restrict__ src, const int* __restrict__ dst,
                     const float* __restrict__ h, float* __restrict__ agg, int E)
{
    int t = blockIdx.x * 256 + threadIdx.x;
    int e = t >> 6;
    if (e >= E) return;
    int q = (t & 63) * 4;
    const float* hs = h   + (size_t)src[e] * DDIM + q;
    float*       ad = agg + (size_t)dst[e] * DDIM + q;
    float4 v = *(const float4*)hs;
    atomicAdd(ad + 0, v.x);
    atomicAdd(ad + 1, v.y);
    atomicAdd(ad + 2, v.z);
    atomicAdd(ad + 3, v.w);
}

// ---------------------------------------------------------------------------
// GCN tail: out = l2norm(agg * inv_in[row] + b).  One wave32 per row.
// ---------------------------------------------------------------------------
__global__ __launch_bounds__(256)
void gcn_norm_kernel(const float* __restrict__ agg, const float* __restrict__ inv_in,
                     const float* __restrict__ b, float* __restrict__ out, int rows)
{
    int wave = threadIdx.x >> 5, lane = threadIdx.x & 31;
    int row = blockIdx.x * 8 + wave;
    if (row >= rows) return;
    float s = inv_in[row];
    const float* a = agg + (size_t)row * DDIM;
    float v[8]; float ss = 0.0f;
    #pragma unroll
    for (int j = 0; j < 8; ++j) {
        int c = lane + j * 32;
        v[j] = a[c] * s + b[c];
        ss += v[j] * v[j];
    }
    #pragma unroll
    for (int m = 16; m >= 1; m >>= 1) ss += __shfl_xor(ss, m, 32);
    float inv = 1.0f / fmaxf(sqrtf(ss), 1e-12f);
    float* o = out + (size_t)row * DDIM;
    #pragma unroll
    for (int j = 0; j < 8; ++j) o[lane + j * 32] = v[j] * inv;
}

// ---------------------------------------------------------------------------
// In-place row l2norm on bf16 buffer (projector output)
// ---------------------------------------------------------------------------
__global__ __launch_bounds__(256)
void rownorm_bf16_kernel(__bf16* __restrict__ buf, int rows)
{
    int wave = threadIdx.x >> 5, lane = threadIdx.x & 31;
    int row = blockIdx.x * 8 + wave;
    if (row >= rows) return;
    __bf16* p = buf + (size_t)row * DDIM;
    float v[8]; float ss = 0.0f;
    #pragma unroll
    for (int j = 0; j < 8; ++j) {
        v[j] = (float)p[lane + j * 32];
        ss += v[j] * v[j];
    }
    #pragma unroll
    for (int m = 16; m >= 1; m >>= 1) ss += __shfl_xor(ss, m, 32);
    float inv = 1.0f / fmaxf(sqrtf(ss), 1e-12f);
    #pragma unroll
    for (int j = 0; j < 8; ++j) p[lane + j * 32] = (__bf16)(v[j] * inv);
}

// ---------------------------------------------------------------------------
// BatchNorm (training stats) over 250K rows x 256 cols
// ---------------------------------------------------------------------------
__global__ __launch_bounds__(256)
void bn_stats_kernel(const __bf16* __restrict__ x, float* __restrict__ sum,
                     float* __restrict__ sumsq, int M)
{
    int col = threadIdx.x;
    float s = 0.0f, s2 = 0.0f;
    for (int r = blockIdx.x; r < M; r += gridDim.x) {
        float v = (float)x[(size_t)r * DDIM + col];
        s += v; s2 += v * v;
    }
    atomicAdd(&sum[col], s);
    atomicAdd(&sumsq[col], s2);
}

__global__ __launch_bounds__(256)
void bn_final_kernel(const float* __restrict__ sum, const float* __restrict__ sumsq,
                     float* __restrict__ mu, float* __restrict__ rstd, float invM)
{
    int c = threadIdx.x;
    float m = sum[c] * invM;
    float v = sumsq[c] * invM - m * m;   // biased var
    mu[c]   = m;
    rstd[c] = rsqrtf(v + BN_EPS_);
}

__global__ __launch_bounds__(256)
void bn_relu_kernel(__bf16* __restrict__ x, const float* __restrict__ mu,
                    const float* __restrict__ rstd, const float* __restrict__ gamma,
                    const float* __restrict__ beta, size_t total)
{
    size_t i = (size_t)blockIdx.x * 256 + threadIdx.x;
    if (i >= total) return;
    int c = (int)(i & (DDIM - 1));
    float v = ((float)x[i] - mu[c]) * rstd[c] * gamma[c] + beta[c];
    x[i] = (__bf16)fmaxf(v, 0.0f);
}

// ---------------------------------------------------------------------------
// Contrastive loss: one wave32 per node (35 dots of length 256 each)
// ---------------------------------------------------------------------------
__global__ __launch_bounds__(256)
void loss_kernel(const float* __restrict__ gnn, const __bf16* __restrict__ pos,
                 const float* __restrict__ hneg, const int* __restrict__ negind,
                 float* __restrict__ accum, int Nn)
{
    int wave = threadIdx.x >> 5, lane = threadIdx.x & 31;
    int n = blockIdx.x * 8 + wave;
    if (n >= Nn) return;

    float g[8], p[SSAMP][8], q[SSAMP][8];
    const float* gr = gnn + (size_t)n * DDIM;
    #pragma unroll
    for (int j = 0; j < 8; ++j) g[j] = gr[lane + j * 32];
    #pragma unroll
    for (int s = 0; s < SSAMP; ++s) {
        const __bf16* pr = pos + ((size_t)n * SSAMP + s) * DDIM;
        #pragma unroll
        for (int j = 0; j < 8; ++j) p[s][j] = (float)pr[lane + j * 32];
        int idx = negind[n * SSAMP + s];
        const float* nr = hneg + (size_t)idx * DDIM;
        #pragma unroll
        for (int j = 0; j < 8; ++j) q[s][j] = nr[lane + j * 32];
    }
    float pd[SSAMP] = {}, nd[SSAMP] = {}, cd[SSAMP][SSAMP] = {};
    #pragma unroll
    for (int s = 0; s < SSAMP; ++s) {
        #pragma unroll
        for (int j = 0; j < 8; ++j) { pd[s] += g[j] * p[s][j]; nd[s] += g[j] * q[s][j]; }
        #pragma unroll
        for (int t = 0; t < SSAMP; ++t)
            #pragma unroll
            for (int j = 0; j < 8; ++j) cd[s][t] += p[s][j] * q[t][j];
    }
    #pragma unroll
    for (int m = 16; m >= 1; m >>= 1) {
        #pragma unroll
        for (int s = 0; s < SSAMP; ++s) {
            pd[s] += __shfl_xor(pd[s], m, 32);
            nd[s] += __shfl_xor(nd[s], m, 32);
            #pragma unroll
            for (int t = 0; t < SSAMP; ++t) cd[s][t] += __shfl_xor(cd[s][t], m, 32);
        }
    }
    if (lane == 0) {
        float pos_score = 0.0f, neg1 = 0.0f;
        #pragma unroll
        for (int s = 0; s < SSAMP; ++s) {
            pos_score += pd[s] * TEMP_INV;       // sum log(exp(x)) == sum x
            neg1 += expf(nd[s] * TEMP_INV);
        }
        float neg_score = 0.0f;
        #pragma unroll
        for (int s = 0; s < SSAMP; ++s) {
            float neg2 = 0.0f;
            #pragma unroll
            for (int t = 0; t < SSAMP; ++t) neg2 += expf(cd[s][t] * TEMP_INV);
            neg_score += logf(LAM_ * neg1 + neg2);
        }
        atomicAdd(accum, neg_score - pos_score); // LAMBDA_LOSS == 1
    }
}

__global__ void finalize_kernel(const float* __restrict__ accum, float* __restrict__ out) {
    if (threadIdx.x == 0 && blockIdx.x == 0)
        out[0] = accum[0] / ((float)NNODES * (float)SSAMP);
}

// ---------------------------------------------------------------------------
// Host orchestration
// ---------------------------------------------------------------------------
extern "C" void kernel_launch(void* const* d_in, const int* in_sizes, int n_in,
                              void* d_out, int out_size, void* d_ws, size_t ws_size,
                              hipStream_t stream)
{
    (void)in_sizes; (void)n_in; (void)out_size; (void)ws_size;

    const float* feat     = (const float*)d_in[0];
    const float* feat_neg = (const float*)d_in[1];
    const float* feat_nei = (const float*)d_in[2];
    const int*   src      = (const int*)d_in[3];
    const int*   dst      = (const int*)d_in[4];
    const int*   src_neg  = (const int*)d_in[5];
    const int*   dst_neg  = (const int*)d_in[6];
    const int*   negind   = (const int*)d_in[7];
    const float* W_gcn    = (const float*)d_in[8];
    const float* b_gcn    = (const float*)d_in[9];
    const float* W1       = (const float*)d_in[10];
    const float* b1       = (const float*)d_in[11];
    const float* gamma    = (const float*)d_in[12];
    const float* beta     = (const float*)d_in[13];
    const float* W2       = (const float*)d_in[14];
    const float* b2       = (const float*)d_in[15];
    const float* Wp       = (const float*)d_in[16];
    const float* bp       = (const float*)d_in[17];

    // ---- workspace carve-out (all 256B aligned) ----
    size_t off = 0;
    char* base = (char*)d_ws;
    auto carve = [&](size_t bytes) -> void* {
        void* p = base + off;
        off = (off + bytes + 255) & ~(size_t)255;
        return p;
    };
    float*  inv_out1 = (float*)carve(NNODES * 4);
    float*  inv_in1  = (float*)carve(NNODES * 4);
    float*  inv_out2 = (float*)carve(NNODES * 4);
    float*  inv_in2  = (float*)carve(NNODES * 4);
    float*  bn_sum   = (float*)carve(DDIM * 4);
    float*  bn_sq    = (float*)carve(DDIM * 4);
    float*  bn_mu    = (float*)carve(DDIM * 4);
    float*  bn_rstd  = (float*)carve(DDIM * 4);
    float*  lacc     = (float*)carve(256);
    __bf16* Wg_bf    = (__bf16*)carve((size_t)DDIM * DDIM * 2);   // transposed
    __bf16* W1_bf    = (__bf16*)carve((size_t)DDIM * DDIM * 2);   // transposed
    __bf16* W2_bf    = (__bf16*)carve((size_t)DDIM * DDIM * 2);   // transposed
    __bf16* Wp_bf    = (__bf16*)carve((size_t)DDIM * DDIM * 2);   // transposed
    float*  hbuf     = (float*)carve((size_t)NNODES * DDIM * 4);
    float*  aggbuf   = (float*)carve((size_t)NNODES * DDIM * 4);
    float*  gnn      = (float*)carve((size_t)NNODES * DDIM * 4);
    float*  hneg     = (float*)carve((size_t)NNODES * DDIM * 4);
    __bf16* tA       = (__bf16*)carve((size_t)NS_ * DDIM * 2);
    __bf16* tB       = (__bf16*)carve((size_t)NS_ * DDIM * 2);

    const int gE   = (NEDGES + 255) / 256;
    const int gN   = (NNODES + 255) / 256;
    const int gGM1 = (NNODES + 63) / 64;           // GEMM blocks, M=N
    const int gGM2 = (NS_ + 63) / 64;              // GEMM blocks, M=N*S
    const int gRowN  = (NNODES + 7) / 8;           // 8 rows (waves) per block
    const int gRowNS = (NS_ + 7) / 8;
    const int gElemNS = (int)(((size_t)NS_ * DDIM + 255) / 256);
    const int gAgg = (NEDGES * 64 + 255) / 256;

    // ---- weights -> bf16, transposed (stay L2-hot for every GEMM block) ----
    f32_to_bf16_T_kernel<<<DDIM, 256, 0, stream>>>(W_gcn, Wg_bf);
    f32_to_bf16_T_kernel<<<DDIM, 256, 0, stream>>>(W1,    W1_bf);
    f32_to_bf16_T_kernel<<<DDIM, 256, 0, stream>>>(W2,    W2_bf);
    f32_to_bf16_T_kernel<<<DDIM, 256, 0, stream>>>(Wp,    Wp_bf);

    // ---- degree normalizers for both graphs ----
    hipMemsetAsync(inv_out1, 0, NNODES * 4, stream);
    hipMemsetAsync(inv_in1,  0, NNODES * 4, stream);
    hipMemsetAsync(inv_out2, 0, NNODES * 4, stream);
    hipMemsetAsync(inv_in2,  0, NNODES * 4, stream);
    hipMemsetAsync(bn_sum,   0, DDIM * 4, stream);
    hipMemsetAsync(bn_sq,    0, DDIM * 4, stream);
    hipMemsetAsync(lacc,     0, 4, stream);
    deg_count_kernel<<<gE, 256, 0, stream>>>(src, dst, inv_out1, inv_in1, NEDGES);
    deg_count_kernel<<<gE, 256, 0, stream>>>(src_neg, dst_neg, inv_out2, inv_in2, NEDGES);
    deg_inv_kernel<<<gN, 256, 0, stream>>>(inv_out1, NNODES);
    deg_inv_kernel<<<gN, 256, 0, stream>>>(inv_in1,  NNODES);
    deg_inv_kernel<<<gN, 256, 0, stream>>>(inv_out2, NNODES);
    deg_inv_kernel<<<gN, 256, 0, stream>>>(inv_in2,  NNODES);

    // ---- GCN on positive graph: gnn = l2norm(agg(h)*inv_in + b) ----
    gemm256_kernel<false, false, false, true><<<gGM1, 256, 0, stream>>>(
        feat, inv_out1, Wg_bf, nullptr, hbuf, NNODES);
    hipMemsetAsync(aggbuf, 0, (size_t)NNODES * DDIM * 4, stream);
    edge_agg_kernel<<<gAgg, 256, 0, stream>>>(src, dst, hbuf, aggbuf, NEDGES);
    gcn_norm_kernel<<<gRowN, 256, 0, stream>>>(aggbuf, inv_in1, b_gcn, gnn, NNODES);

    // ---- GCN on negative graph: h_neg ----
    gemm256_kernel<false, false, false, true><<<gGM1, 256, 0, stream>>>(
        feat_neg, inv_out2, Wg_bf, nullptr, hbuf, NNODES);
    hipMemsetAsync(aggbuf, 0, (size_t)NNODES * DDIM * 4, stream);
    edge_agg_kernel<<<gAgg, 256, 0, stream>>>(src_neg, dst_neg, hbuf, aggbuf, NEDGES);
    gcn_norm_kernel<<<gRowN, 256, 0, stream>>>(aggbuf, inv_in2, b_gcn, hneg, NNODES);

    // ---- target MLP: tA = relu(BN(nei @ W1 + b1)); tB = tA @ W2 + b2 ----
    gemm256_kernel<false, true, true, false><<<gGM2, 256, 0, stream>>>(
        feat_nei, nullptr, W1_bf, b1, tA, NS_);
    bn_stats_kernel<<<1024, 256, 0, stream>>>(tA, bn_sum, bn_sq, NS_);
    bn_final_kernel<<<1, 256, 0, stream>>>(bn_sum, bn_sq, bn_mu, bn_rstd, 1.0f / (float)NS_);
    bn_relu_kernel<<<gElemNS, 256, 0, stream>>>(tA, bn_mu, bn_rstd, gamma, beta,
                                                (size_t)NS_ * DDIM);
    gemm256_kernel<true, true, true, false><<<gGM2, 256, 0, stream>>>(
        tA, nullptr, W2_bf, b2, tB, NS_);

    // ---- projector + l2norm: pos_list (in tA) ----
    gemm256_kernel<true, true, true, false><<<gGM2, 256, 0, stream>>>(
        tB, nullptr, Wp_bf, bp, tA, NS_);
    rownorm_bf16_kernel<<<gRowNS, 256, 0, stream>>>(tA, NS_);

    // ---- contrastive loss reduction ----
    loss_kernel<<<gRowN, 256, 0, stream>>>(gnn, tA, hneg, negind, lacc, NNODES);
    finalize_kernel<<<1, 1, 0, stream>>>(lacc, (float*)d_out);
}
